// PN2GeometryEncoder_6734508720335
// MI455X (gfx1250) — matrix operand order repaired and checked by reading, not compiled, not run on hardware
//
#include <hip/hip_runtime.h>

// ---------------------------------------------------------------------------
// Types for CDNA5 WMMA (wave32). bf16 chosen: f32 inputs keep exponent range.
// ---------------------------------------------------------------------------
typedef __bf16 bf16_t;
typedef bf16_t bf16x16 __attribute__((ext_vector_type(16)));
typedef float  f32x8   __attribute__((ext_vector_type(8)));

__device__ __forceinline__ unsigned short f2bf(float f) {
    union { float f; unsigned int u; } c; c.f = f;
    unsigned int u = c.u;
    u += 0x7fffu + ((u >> 16) & 1u);            // round-to-nearest-even
    return (unsigned short)(u >> 16);
}

// ---------------------------------------------------------------------------
// Problem constants (match reference)
// ---------------------------------------------------------------------------
constexpr int cB  = 16;
constexpr int cN  = 8192;
constexpr int cN1 = 512;
constexpr int cN2 = 128;
constexpr int cK1 = 32;
constexpr int cK2 = 64;
constexpr int cCG = 256;

// ---------------------------------------------------------------------------
// WMMA GEMM:  Y[M,Nc] = A[M,0:K] @ W[K,Nc] + bias   (bf16 x bf16 -> f32)
// A is stored with padded row stride lda (multiple of 32, zero-filled pad), so
// the K-loop runs over lda and A staging is always full tiles.
// 128x64 tile per 256-thread block (8 wave32s). Wave w owns rows w*16..w*16+15
// and all 64 columns: 1 A fragment, 4 B fragments, 4 WMMAs per K-step.
// LDS: A row-major [m][k]; W tile transposed to n-major [n][k] so every
// fragment dword is one aligned ds_load_b32 (pairs merge to b64/b128).
// ---------------------------------------------------------------------------
__global__ __launch_bounds__(256)
void wmma_gemm_bias(const float* __restrict__ A, const float* __restrict__ W,
                    const float* __restrict__ bias, float* __restrict__ Y,
                    int M, int K, int lda, int Nc)
{
    __shared__ unsigned short As[128][36];  // [m][k], +4 pad
    __shared__ unsigned short Bs[64][40];   // [n][k], +8 pad

    const int tid  = threadIdx.x;
    const int wave = tid >> 5;
    const int lane = tid & 31;
    const int hh   = lane >> 4;         // lane half (ISA fragment layout)
    const int lr   = lane & 15;
    const int row0 = blockIdx.y * 128;
    const int col0 = blockIdx.x * 64;

    const bool fastA = (row0 + 128 <= M);   // lda % 32 == 0 by construction

    f32x8 acc0 = {}, acc1 = {}, acc2 = {}, acc3 = {};

    for (int kk = 0; kk < lda; kk += 32) {
        // ---- stage A tile: 128x32 ----
        if (fastA) {
            #pragma unroll
            for (int it = 0; it < 4; ++it) {
                int e = tid + it * 256;          // 0..1023 float4s
                int r = e >> 3;                  // 0..127
                int c = (e & 7) << 2;            // 0,4,..,28
                const float4 v = *(const float4*)(A + (size_t)(row0 + r) * lda + kk + c);
                unsigned long long p =
                      (unsigned long long)((unsigned)f2bf(v.x) | ((unsigned)f2bf(v.y) << 16))
                    | ((unsigned long long)((unsigned)f2bf(v.z) | ((unsigned)f2bf(v.w) << 16)) << 32);
                *(unsigned long long*)&As[r][c] = p;
            }
        } else {
            for (int e = tid; e < 128 * 32; e += 256) {
                int r = e >> 5, c = e & 31;
                int gr = row0 + r;
                float v = (gr < M) ? A[(size_t)gr * lda + kk + c] : 0.0f;
                As[r][c] = f2bf(v);
            }
        }
        // ---- stage W tile 32x64, transposed to n-major ----
        if (kk + 32 <= K) {
            #pragma unroll
            for (int it = 0; it < 2; ++it) {
                int e = tid + it * 256;          // 0..511 float4s
                int r = e >> 4;                  // k row 0..31
                int c = (e & 15) << 2;           // n 0..60
                const float4 v = *(const float4*)(W + (size_t)(kk + r) * Nc + col0 + c);
                Bs[c + 0][r] = f2bf(v.x);
                Bs[c + 1][r] = f2bf(v.y);
                Bs[c + 2][r] = f2bf(v.z);
                Bs[c + 3][r] = f2bf(v.w);
            }
        } else {
            for (int e = tid; e < 32 * 64; e += 256) {
                int r = e >> 6, c = e & 63;
                int gr = kk + r;
                float v = (gr < K) ? W[(size_t)gr * Nc + col0 + c] : 0.0f;
                Bs[c][r] = f2bf(v);
            }
        }
        // speculative prefetch of next A K-tile -> global_prefetch_b8
        if (kk + 32 < lda) {
            const float* pf = A + (size_t)(row0 + (tid >> 1)) * lda + (kk + 32);
            __builtin_prefetch(pf, 0, 0);
        }
        __syncthreads();

        union Frag { bf16x16 v; unsigned int u[8]; };
        Frag a, b0, b1, b2, b3;
        // A 16x32 bf16 fragment: lanes 0-15 row M=lr hold
        // K {0..15}; lanes 16-31 hold K {16..31}; pairs per dword.
        int arow = wave * 16 + lr;
        #pragma unroll
        for (int vg = 0; vg < 8; ++vg) {
            int ka = 2 * vg + 16 * hh;
            a.u[vg] = *(const unsigned int*)&As[arow][ka];
        }
        // B 32x16 fragments: col = lane%16, dword vg holds K {2vg,2vg+1}+16*half.
        #pragma unroll
        for (int vg = 0; vg < 8; ++vg) {
            int kb = 2 * vg + 16 * hh;
            b0.u[vg] = *(const unsigned int*)&Bs[ 0 + lr][kb];
            b1.u[vg] = *(const unsigned int*)&Bs[16 + lr][kb];
            b2.u[vg] = *(const unsigned int*)&Bs[32 + lr][kb];
            b3.u[vg] = *(const unsigned int*)&Bs[48 + lr][kb];
        }
        acc0 = __builtin_amdgcn_wmma_f32_16x16x32_bf16(false, a.v, false, b0.v, (short)0, acc0, false, false);
        acc1 = __builtin_amdgcn_wmma_f32_16x16x32_bf16(false, a.v, false, b1.v, (short)0, acc1, false, false);
        acc2 = __builtin_amdgcn_wmma_f32_16x16x32_bf16(false, a.v, false, b2.v, (short)0, acc2, false, false);
        acc3 = __builtin_amdgcn_wmma_f32_16x16x32_bf16(false, a.v, false, b3.v, (short)0, acc3, false, false);
        __syncthreads();
    }

    // C/D layout: VGPR r -> M = r + 8*half, N = lane%16
    f32x8* accs[4] = { &acc0, &acc1, &acc2, &acc3 };
    #pragma unroll
    for (int bt = 0; bt < 4; ++bt) {
        int n = col0 + bt * 16 + lr;        // n < Nc (Nc multiple of 64, exact grid)
        float bz = bias[n];
        #pragma unroll
        for (int r = 0; r < 8; ++r) {
            int m = row0 + wave * 16 + r + 8 * hh;
            if (m < M) Y[(size_t)m * Nc + n] = (*accs[bt])[r] + bz;
        }
    }
}

// ---------------------------------------------------------------------------
// Masked per-column mean/var (training-mode BN stats). One block per column.
// ---------------------------------------------------------------------------
__global__ __launch_bounds__(256)
void column_stats(const float* __restrict__ Y, const int* __restrict__ mask,
                  float* __restrict__ stats, int M, int Nc)
{
    __shared__ float ss[256], sq[256], sc[256];
    int c = blockIdx.x, tid = threadIdx.x;
    float s = 0.f, q = 0.f, n = 0.f;
    for (int r = tid; r < M; r += 256) {
        if (!mask || mask[r]) {
            float v = Y[(size_t)r * Nc + c];
            s += v; q += v * v; n += 1.0f;
        }
    }
    ss[tid] = s; sq[tid] = q; sc[tid] = n;
    __syncthreads();
    for (int o = 128; o > 0; o >>= 1) {
        if (tid < o) { ss[tid] += ss[tid + o]; sq[tid] += sq[tid + o]; sc[tid] += sc[tid + o]; }
        __syncthreads();
    }
    if (tid == 0) {
        float cnt  = fmaxf(sc[0], 1.0f);
        float mean = ss[0] / cnt;
        float var  = sq[0] / cnt - mean * mean;
        stats[c] = mean; stats[Nc + c] = var;
    }
}

__global__ void bn_relu(float* __restrict__ Y, const float* __restrict__ stats,
                        const float* __restrict__ gamma, const float* __restrict__ beta,
                        int M, int Nc)
{
    size_t t = (size_t)blockIdx.x * blockDim.x + threadIdx.x;
    if (t >= (size_t)M * Nc) return;
    int c = (int)(t % Nc);
    float v = (Y[t] - stats[c]) * rsqrtf(stats[Nc + c] + 1e-5f) * gamma[c] + beta[c];
    Y[t] = fmaxf(v, 0.0f);
}

// ---------------------------------------------------------------------------
// Farthest point sampling: one block per cloud, dist array resident in LDS.
// ---------------------------------------------------------------------------
__global__ __launch_bounds__(1024)
void fps_kernel(const float* __restrict__ pos, int* __restrict__ out_idx,
                int Npts, int nsample)
{
    __shared__ float dist[8192];
    __shared__ float rv[1024];
    __shared__ int   ri[1024];
    __shared__ int   s_next;
    const float* p = pos + (size_t)blockIdx.x * Npts * 3;
    int tid = threadIdx.x;
    float p0x = p[0], p0y = p[1], p0z = p[2];
    for (int i = tid; i < Npts; i += 1024) {
        float dx = p[i*3] - p0x, dy = p[i*3+1] - p0y, dz = p[i*3+2] - p0z;
        dist[i] = dx*dx + dy*dy + dz*dz;
    }
    if (tid == 0) out_idx[(size_t)blockIdx.x * nsample] = 0;
    for (int s = 1; s < nsample; ++s) {
        __syncthreads();
        float bv = -1.0f; int bi = 0;
        for (int i = tid; i < Npts; i += 1024) { float d = dist[i]; if (d > bv) { bv = d; bi = i; } }
        rv[tid] = bv; ri[tid] = bi;
        __syncthreads();
        for (int o = 512; o > 0; o >>= 1) {
            if (tid < o) {
                if (rv[tid+o] > rv[tid] || (rv[tid+o] == rv[tid] && ri[tid+o] < ri[tid])) {
                    rv[tid] = rv[tid+o]; ri[tid] = ri[tid+o];
                }
            }
            __syncthreads();
        }
        if (tid == 0) { s_next = ri[0]; out_idx[(size_t)blockIdx.x * nsample + s] = ri[0]; }
        __syncthreads();
        int nx = s_next;
        float nxx = p[nx*3], nxy = p[nx*3+1], nxz = p[nx*3+2];
        for (int i = tid; i < Npts; i += 1024) {
            float dx = p[i*3] - nxx, dy = p[i*3+1] - nxy, dz = p[i*3+2] - nxz;
            float d = dx*dx + dy*dy + dz*dz;
            if (d < dist[i]) dist[i] = d;
        }
    }
}

__global__ void gather3(const float* __restrict__ pos, const int* __restrict__ idx,
                        float* __restrict__ out, int Npts, int n)
{
    int t = blockIdx.x * blockDim.x + threadIdx.x;
    if (t >= cB * n * 3) return;
    int b = t / (n * 3); int rem = t % (n * 3); int i = rem / 3; int d = rem % 3;
    out[t] = pos[((size_t)b * Npts + idx[b * n + i]) * 3 + d];
}

// ---------------------------------------------------------------------------
// Ball query: one wave32 per center; lowest-index-first selection via ballot.
// ---------------------------------------------------------------------------
__global__ void ball_group(const float* __restrict__ src, const float* __restrict__ ctr,
                           int* __restrict__ oidx, int* __restrict__ ovalid,
                           int Nsrc, int Nctr, float r2, int k, int nwaves)
{
    int g = blockIdx.x * blockDim.x + threadIdx.x;
    int w = g >> 5, lane = g & 31;
    if (w >= nwaves) return;
    int b = w / Nctr, c = w % Nctr;
    const float* cp = ctr + (size_t)(b * Nctr + c) * 3;
    float cx = cp[0], cy = cp[1], cz = cp[2];
    const float* sp = src + (size_t)b * Nsrc * 3;
    size_t obase = (size_t)(b * Nctr + c) * k;
    int cnt = 0;
    for (int base = 0; base < Nsrc && cnt < k; base += 32) {
        int i = base + lane;
        bool ok = false;
        if (i < Nsrc) {
            float dx = sp[i*3] - cx, dy = sp[i*3+1] - cy, dz = sp[i*3+2] - cz;
            ok = (dx*dx + dy*dy + dz*dz) <= r2;
        }
        unsigned int m = (unsigned int)__ballot(ok);   // wave32: low 32 bits
        while (m && cnt < k) {
            int j = __ffs(m) - 1;
            m &= m - 1;
            if (lane == 0) { oidx[obase + cnt] = base + j; ovalid[obase + cnt] = 1; }
            ++cnt;
        }
    }
    if (lane == 0)
        for (int t = cnt; t < k; ++t) { oidx[obase + t] = 0; ovalid[obase + t] = 0; }
}

// Build PointNetConv input rows: [x_j | pos_j - pos_i | 0-pad], row stride ldout
__global__ void build_sa_input(const float* __restrict__ feat, const float* __restrict__ srcpos,
                               const float* __restrict__ ctrpos, const int* __restrict__ idx,
                               float* __restrict__ X, int Nsrc, int Nctr, int k, int C, int ldout)
{
    size_t t = (size_t)blockIdx.x * blockDim.x + threadIdx.x;
    size_t total = (size_t)cB * Nctr * k * ldout;
    if (t >= total) return;
    size_t r = t / ldout; int c = (int)(t % ldout);
    float v = 0.0f;
    if (c < C + 3) {
        int b = (int)(r / ((size_t)Nctr * k));
        size_t rem = r % ((size_t)Nctr * k);
        int i = (int)(rem / k);
        int s = idx[r];
        if (c < C) v = feat[((size_t)b * Nsrc + s) * C + c];
        else { int d = c - C; v = srcpos[((size_t)b*Nsrc + s)*3 + d] - ctrpos[((size_t)b*Nctr + i)*3 + d]; }
    }
    X[t] = v;
}

__global__ void masked_max(const float* __restrict__ H, const int* __restrict__ valid,
                           float* __restrict__ out, int Rows, int k, int C)
{
    size_t t = (size_t)blockIdx.x * blockDim.x + threadIdx.x;
    if (t >= (size_t)Rows * C) return;
    size_t r = t / C; int c = (int)(t % C);
    size_t base = r * k;
    float best = -3.4e38f;
    for (int j = 0; j < k; ++j)
        if (valid[base + j]) best = fmaxf(best, H[(base + j) * C + c]);
    out[t] = best;
}

__global__ void rowmax(const float* __restrict__ X, float* __restrict__ out,
                       int Bn, int Nr, int C)
{
    int t = blockIdx.x * blockDim.x + threadIdx.x;
    if (t >= Bn * C) return;
    int b = t / C, c = t % C;
    float best = -3.4e38f;
    for (int i = 0; i < Nr; ++i)
        best = fmaxf(best, X[((size_t)b * Nr + i) * C + c]);
    out[t] = best;
}

// 3-NN search (thread per target, running sorted top-3)
__global__ void knn3(const float* __restrict__ tgt, const float* __restrict__ srcp,
                     int* __restrict__ oi, float* __restrict__ od, int Bn, int Nt, int Ns)
{
    int t = blockIdx.x * blockDim.x + threadIdx.x;
    if (t >= Bn * Nt) return;
    int b = t / Nt;
    const float* tp = tgt + (size_t)t * 3;
    float tx = tp[0], ty = tp[1], tz = tp[2];
    const float* sp = srcp + (size_t)b * Ns * 3;
    float d0 = 3.4e38f, d1 = 3.4e38f, d2v = 3.4e38f; int i0 = 0, i1 = 0, i2 = 0;
    for (int i = 0; i < Ns; ++i) {
        float dx = sp[i*3] - tx, dy = sp[i*3+1] - ty, dz = sp[i*3+2] - tz;
        float d = dx*dx + dy*dy + dz*dz;
        if      (d < d0)  { d2v = d1; i2 = i1; d1 = d0; i1 = i0; d0 = d;  i0 = i; }
        else if (d < d1)  { d2v = d1; i2 = i1; d1 = d;  i1 = i; }
        else if (d < d2v) { d2v = d;  i2 = i; }
    }
    oi[t*3] = i0; oi[t*3+1] = i1; oi[t*3+2] = i2;
    od[t*3] = d0; od[t*3+1] = d1; od[t*3+2] = d2v;
}

__global__ void knn_interp_apply(const float* __restrict__ feats, const int* __restrict__ oi,
                                 const float* __restrict__ od, float* __restrict__ out,
                                 int Bn, int Nt, int Ns, int C)
{
    size_t t = (size_t)blockIdx.x * blockDim.x + threadIdx.x;
    if (t >= (size_t)Bn * Nt * C) return;
    size_t r = t / C; int c = (int)(t % C);
    int b = (int)(r / Nt);
    float acc = 0.f, wsum = 0.f;
    #pragma unroll
    for (int j = 0; j < 3; ++j) {
        float w = 1.0f / fmaxf(od[r*3 + j], 1e-16f);
        int i = oi[r*3 + j];
        acc  += w * feats[((size_t)b * Ns + i) * C + c];
        wsum += w;
    }
    out[t] = acc / wsum;
}

// Concatenate [X1 | X2 | 0-pad], row stride ldout
__global__ void concat2(const float* __restrict__ X1, const float* __restrict__ X2,
                        float* __restrict__ out, int Rows, int C1, int C2, int ldout)
{
    size_t t = (size_t)blockIdx.x * blockDim.x + threadIdx.x;
    if (t >= (size_t)Rows * ldout) return;
    size_t r = t / ldout; int c = (int)(t % ldout);
    float v = 0.0f;
    if (c < C1) v = X1[r * C1 + c];
    else if (c < C1 + C2) v = X2[r * C2 + (c - C1)];
    out[t] = v;
}

__global__ void copyk(const float* __restrict__ src, float* __restrict__ dst, size_t n)
{
    size_t t = (size_t)blockIdx.x * blockDim.x + threadIdx.x;
    if (t < n) dst[t] = src[t];
}

// ---------------------------------------------------------------------------
// Host orchestration
// ---------------------------------------------------------------------------
static inline unsigned EW(size_t n) { return (unsigned)((n + 255) / 256); }

static void mlp_layer(const float* X, const float* W, const float* bvec,
                      const float* gamma, const float* beta, const int* mask,
                      float* Y, float* stats, int M, int K, int lda, int Nc,
                      hipStream_t s)
{
    dim3 grid((Nc + 63) / 64, (M + 127) / 128);
    wmma_gemm_bias<<<grid, 256, 0, s>>>(X, W, bvec, Y, M, K, lda, Nc);
    column_stats<<<Nc, 256, 0, s>>>(Y, mask, stats, M, Nc);
    bn_relu<<<EW((size_t)M * Nc), 256, 0, s>>>(Y, stats, gamma, beta, M, Nc);
}

extern "C" void kernel_launch(void* const* d_in, const int* in_sizes, int n_in,
                              void* d_out, int out_size, void* d_ws, size_t ws_size,
                              hipStream_t stream)
{
    (void)in_sizes; (void)n_in; (void)out_size; (void)ws_size;
    const float* pts = (const float*)d_in[0];
    auto P = [&](int i) { return (const float*)d_in[i]; };
    // param bases: each layer = {W,b,gamma,beta}
    // sa1_local: 1,5,9  sa1_global: 13  sa2_local: 17,21,25  sa2_global: 29
    // glob: 33,37  fp1: 41,45  fp0: 49,53

    float* w  = (float*)d_ws;
    int*   wi = (int*)d_ws;

    constexpr size_t OFF_IDX1  = 0;           // 8192 int
    constexpr size_t OFF_POS1  = 8192;        // 24576
    constexpr size_t OFF_IDX2  = 32768;       // 2048 int
    constexpr size_t OFF_POS2  = 34816;       // 6144
    constexpr size_t OFF_GI1   = 40960;       // 262144 int
    constexpr size_t OFF_GV1   = 303104;      // 262144 int
    constexpr size_t OFF_GI2   = 565248;      // 131072 int
    constexpr size_t OFF_GV2   = 696320;      // 131072 int
    constexpr size_t OFF_MX1   = 827392;      // 1048576
    constexpr size_t OFF_X1    = 1875968;     // 2097152
    constexpr size_t OFF_MX2   = 3973120;     // 524288
    constexpr size_t OFF_X2    = 4497408;     // 524288
    constexpr size_t OFF_XG    = 5021696;     // 4096
    constexpr size_t OFF_GMID  = 5025792;     // 8192
    constexpr size_t OFF_GOUT  = 5033984;     // 4096
    constexpr size_t OFF_X1UP  = 5038080;     // 2097152
    constexpr size_t OFF_CAT1  = 7135232;     // 4194304
    constexpr size_t OFF_F1A   = 11329536;    // 2097152
    constexpr size_t OFF_X1FP  = 13426688;    // 2097152
    constexpr size_t OFF_KI1   = 15523840;    // 24576 int
    constexpr size_t OFF_KD1   = 15548416;    // 24576
    constexpr size_t OFF_KI0   = 15572992;    // 393216 int
    constexpr size_t OFF_KD0   = 15966208;    // 393216
    constexpr size_t OFF_STATS = 16359424;    // 2048
    constexpr size_t OFF_BUFA  = 16361472;    // 37748736 = 131072x288 (max tile)
    constexpr size_t OFF_BUFB  = 54110208;    // 37748736

    float* stats = w + OFF_STATS;
    float* bufA  = w + OFF_BUFA;
    float* bufB  = w + OFF_BUFB;

    // ---------------- SA1 ----------------
    fps_kernel<<<cB, 1024, 0, stream>>>(pts, wi + OFF_IDX1, cN, cN1);
    gather3<<<EW((size_t)cB * cN1 * 3), 256, 0, stream>>>(pts, wi + OFF_IDX1, w + OFF_POS1, cN, cN1);
    ball_group<<<EW((size_t)cB * cN1 * 32), 256, 0, stream>>>(
        pts, w + OFF_POS1, wi + OFF_GI1, wi + OFF_GV1, cN, cN1, 0.2f * 0.2f, cK1, cB * cN1);

    int M1 = cB * cN1 * cK1;   // 262144
    build_sa_input<<<EW((size_t)M1 * 32), 256, 0, stream>>>(
        pts, pts, w + OFF_POS1, wi + OFF_GI1, bufA, cN, cN1, cK1, 3, 32);
    mlp_layer(bufA, P(1),  P(2),  P(3),  P(4),  wi + OFF_GV1, bufB, stats, M1, 6,  32, 64,  stream);
    mlp_layer(bufB, P(5),  P(6),  P(7),  P(8),  wi + OFF_GV1, bufA, stats, M1, 64, 64, 64,  stream);
    mlp_layer(bufA, P(9),  P(10), P(11), P(12), wi + OFF_GV1, bufB, stats, M1, 64, 64, 128, stream);
    masked_max<<<EW((size_t)cB * cN1 * 128), 256, 0, stream>>>(bufB, wi + OFF_GV1, w + OFF_MX1, cB * cN1, cK1, 128);
    mlp_layer(w + OFF_MX1, P(13), P(14), P(15), P(16), nullptr, w + OFF_X1, stats, cB * cN1, 128, 128, 256, stream);

    // ---------------- SA2 ----------------
    fps_kernel<<<cB, 1024, 0, stream>>>(w + OFF_POS1, wi + OFF_IDX2, cN1, cN2);
    gather3<<<EW((size_t)cB * cN2 * 3), 256, 0, stream>>>(w + OFF_POS1, wi + OFF_IDX2, w + OFF_POS2, cN1, cN2);
    ball_group<<<EW((size_t)cB * cN2 * 32), 256, 0, stream>>>(
        w + OFF_POS1, w + OFF_POS2, wi + OFF_GI2, wi + OFF_GV2, cN1, cN2, 0.4f * 0.4f, cK2, cB * cN2);

    int M2 = cB * cN2 * cK2;   // 131072
    build_sa_input<<<EW((size_t)M2 * 288), 256, 0, stream>>>(
        w + OFF_X1, w + OFF_POS1, w + OFF_POS2, wi + OFF_GI2, bufA, cN1, cN2, cK2, 256, 288);
    mlp_layer(bufA, P(17), P(18), P(19), P(20), wi + OFF_GV2, bufB, stats, M2, 259, 288, 128, stream);
    mlp_layer(bufB, P(21), P(22), P(23), P(24), wi + OFF_GV2, bufA, stats, M2, 128, 128, 128, stream);
    mlp_layer(bufA, P(25), P(26), P(27), P(28), wi + OFF_GV2, bufB, stats, M2, 128, 128, 256, stream);
    masked_max<<<EW((size_t)cB * cN2 * 256), 256, 0, stream>>>(bufB, wi + OFF_GV2, w + OFF_MX2, cB * cN2, cK2, 256);
    mlp_layer(w + OFF_MX2, P(29), P(30), P(31), P(32), nullptr, w + OFF_X2, stats, cB * cN2, 256, 256, 256, stream);

    // ---------------- global branch ----------------
    rowmax<<<EW((size_t)cB * 256), 256, 0, stream>>>(w + OFF_X2, w + OFF_XG, cB, cN2, 256);
    mlp_layer(w + OFF_XG,   P(33), P(34), P(35), P(36), nullptr, w + OFF_GMID, stats, cB, 256, 256, 512, stream);
    mlp_layer(w + OFF_GMID, P(37), P(38), P(39), P(40), nullptr, w + OFF_GOUT, stats, cB, 512, 512, 256, stream);
    copyk<<<EW((size_t)cB * cCG), 256, 0, stream>>>(w + OFF_GOUT, (float*)d_out + (size_t)cB * cN * cCG, (size_t)cB * cCG);

    // ---------------- FP1: x2 -> pos1 ----------------
    knn3<<<EW((size_t)cB * cN1), 256, 0, stream>>>(w + OFF_POS1, w + OFF_POS2, wi + OFF_KI1, w + OFF_KD1, cB, cN1, cN2);
    knn_interp_apply<<<EW((size_t)cB * cN1 * 256), 256, 0, stream>>>(
        w + OFF_X2, wi + OFF_KI1, w + OFF_KD1, w + OFF_X1UP, cB, cN1, cN2, 256);
    concat2<<<EW((size_t)cB * cN1 * 512), 256, 0, stream>>>(w + OFF_X1UP, w + OFF_X1, w + OFF_CAT1, cB * cN1, 256, 256, 512);
    mlp_layer(w + OFF_CAT1, P(41), P(42), P(43), P(44), nullptr, w + OFF_F1A,  stats, cB * cN1, 512, 512, 256, stream);
    mlp_layer(w + OFF_F1A,  P(45), P(46), P(47), P(48), nullptr, w + OFF_X1FP, stats, cB * cN1, 256, 256, 256, stream);

    // ---------------- FP0: x1fp -> pts ----------------
    knn3<<<EW((size_t)cB * cN), 256, 0, stream>>>(pts, w + OFF_POS1, wi + OFF_KI0, w + OFF_KD0, cB, cN, cN1);
    knn_interp_apply<<<EW((size_t)cB * cN * 256), 256, 0, stream>>>(
        w + OFF_X1FP, wi + OFF_KI0, w + OFF_KD0, bufA, cB, cN, cN1, 256);
    concat2<<<EW((size_t)cB * cN * 288), 256, 0, stream>>>(bufA, pts, bufB, cB * cN, 256, 3, 288);
    int M0 = cB * cN;          // 131072
    mlp_layer(bufB, P(49), P(50), P(51), P(52), nullptr, bufA,          stats, M0, 259, 288, 256, stream);
    mlp_layer(bufA, P(53), P(54), P(55), P(56), nullptr, (float*)d_out, stats, M0, 256, 256, 256, stream);
}
